// Layer_20521353740333
// MI455X (gfx1250) — compile-verified, matrix operand order checked by reading, not run on hardware
//
#include <hip/hip_runtime.h>
#include <hip/hip_bf16.h>

// ---------------------------------------------------------------------------
// Performer (FAVOR+) layer forward for MI455X (gfx1250, wave32, WMMA).
// B=4, S=1024, D=1024, H=16, dh=64, M=128, DFF=4096.
// GEMMs: bf16 inputs, v_wmma_f32_16x16x32_bf16, async global->LDS staging
// (GLOBAL_LOAD_ASYNC_TO_LDS_B128 + s_wait_asynccnt) with double buffering.
// Scan / phi / layernorm: fp32 VALU with LDS-resident state.
// ---------------------------------------------------------------------------

typedef __attribute__((ext_vector_type(16))) __bf16 v16bf;
typedef __attribute__((ext_vector_type(8)))  float  v8f;
typedef __attribute__((ext_vector_type(4)))  int    v4i;

#define Bc   4
#define Sc   1024
#define Dc   1024
#define Hc   16
#define DHc  64
#define Mc   128
#define DFFc 4096

#if defined(__gfx1250__) && __has_builtin(__builtin_amdgcn_global_load_async_to_lds_b128) && __has_builtin(__builtin_amdgcn_s_wait_asynccnt)
#define ASYNC_BUILTIN 1
#else
#define ASYNC_BUILTIN 0
#endif

__device__ __forceinline__ void async_ld_b128(const void* g, const void* l) {
#if ASYNC_BUILTIN
    __builtin_amdgcn_global_load_async_to_lds_b128(
        (__attribute__((address_space(1))) v4i*)(unsigned long long)(g),
        (__attribute__((address_space(3))) v4i*)(unsigned)(unsigned long long)(l),
        0, 0);
#elif defined(__HIP_DEVICE_COMPILE__) && defined(__gfx1250__)
    unsigned lo = (unsigned)(unsigned long long)l;   // LDS byte offset (addr[31:0])
    asm volatile("global_load_async_to_lds_b128 %0, %1, off"
                 :: "v"(lo), "v"((unsigned long long)g) : "memory");
#else
    (void)g; (void)l;
#endif
}

template<int N>
__device__ __forceinline__ void wait_async() {
#if ASYNC_BUILTIN
    __builtin_amdgcn_s_wait_asynccnt(N);
#elif defined(__HIP_DEVICE_COMPILE__) && defined(__gfx1250__)
    if constexpr (N == 0) asm volatile("s_wait_asynccnt 0x0" ::: "memory");
    else                  asm volatile("s_wait_asynccnt 0x3" ::: "memory");
#endif
}

__device__ __forceinline__ unsigned short f32_to_bf16_rne(float f) {
    unsigned int u = __float_as_uint(f);
    unsigned int r = u + 0x7FFFu + ((u >> 16) & 1u);
    return (unsigned short)(r >> 16);
}

// ---------------------------------------------------------------- convert ---
__global__ __launch_bounds__(256)
void cvt_f32_bf16(const float* __restrict__ in, unsigned short* __restrict__ out, int n) {
    int i = blockIdx.x * 256 + threadIdx.x;
    if (i < n) out[i] = f32_to_bf16_rne(in[i]);
}

// in: fp32 [K,N] -> out: bf16 [N,K]  (weights transposed once so GEMM B-tiles
// are byte-contiguous and async-copyable)
__global__ __launch_bounds__(256)
void cvt_transpose_bf16(const float* __restrict__ in, unsigned short* __restrict__ out,
                        int K, int N)
{
    __shared__ unsigned short tile[32][33];
    const int kb = blockIdx.y * 32, nb = blockIdx.x * 32;
    const int tx = threadIdx.x & 31, ty = threadIdx.x >> 5;   // 8 rows / pass
#pragma unroll
    for (int r = ty; r < 32; r += 8)
        tile[r][tx] = f32_to_bf16_rne(in[(size_t)(kb + r) * N + nb + tx]);
    __syncthreads();
#pragma unroll
    for (int r = ty; r < 32; r += 8)
        out[(size_t)(nb + r) * K + kb + tx] = tile[tx][r];
}

// ------------------------------------------------------------------- GEMM ---
// C[M,N] = A[M,K] * Bt[N,K]^T  (A,Bt bf16 row-major; fp32 accumulate)
// MODE 0: store f32        MODE 1: +bias, relu, store bf16   MODE 2: +bias, f32
template<int MODE>
__global__ __launch_bounds__(256)
void gemm_wmma_bf16(const unsigned short* __restrict__ A,
                    const unsigned short* __restrict__ Bt,
                    const float* __restrict__ bias,
                    void* __restrict__ Cout,
                    int M, int N, int K)
{
    constexpr int BM = 128, BN = 64, BK = 32;
    constexpr int LDA = BK + 8;                   // ushort stride, padded
    constexpr int LDB = BK + 8;
    __shared__ unsigned short sA[2][BM * LDA];
    __shared__ unsigned short sB[2][BN * LDB];

    const int tid   = threadIdx.x;
    const int wave  = tid >> 5;
    const int lane  = tid & 31;
    const int wm    = wave & 3;                   // 4 waves along M
    const int wn    = wave >> 2;                  // 2 waves along N
    const int laneR = lane & 15;
    const int laneH = lane >> 4;
    const int bm0   = blockIdx.y * BM;
    const int bn0   = blockIdx.x * BN;

    v8f c[2][2] = {};

    // per-thread async chunk assignment: 16B chunks, 4 per 32-bf16 row
    const int rA0 = tid >> 2,           qA0 = tid & 3;              // A chunk 0
    const int rA1 = (tid + 256) >> 2,   qA1 = tid & 3;              // A chunk 1
    const int rB  = tid >> 2,           qB  = tid & 3;              // B chunk
    const char* gA0 = (const char*)(A  + ((size_t)(bm0 + rA0) * K + qA0 * 8));
    const char* gA1 = (const char*)(A  + ((size_t)(bm0 + rA1) * K + qA1 * 8));
    const char* gB  = (const char*)(Bt + ((size_t)(bn0 + rB ) * K + qB  * 8));

    const int nT = K / BK;
    auto issue = [&](int t, int buf) {
        size_t go = (size_t)t * (BK * 2);         // byte advance along K
        async_ld_b128(gA0 + go, &sA[buf][rA0 * LDA + qA0 * 8]);
        async_ld_b128(gA1 + go, &sA[buf][rA1 * LDA + qA1 * 8]);
        async_ld_b128(gB  + go, &sB [buf][rB  * LDB + qB  * 8]);
    };

    issue(0, 0);
    for (int t = 0; t < nT; ++t) {
        const int buf = t & 1;
        if (t + 1 < nT) { issue(t + 1, buf ^ 1); wait_async<3>(); }
        else           { wait_async<0>(); }
        __syncthreads();

        // fragments (16-bit A 16x32 layout, ISA 7.12.2; B tile stored K-major)
        union Frag { v16bf v; unsigned int u[8]; };
        Frag af[2], bf2[2];
#pragma unroll
        for (int i = 0; i < 2; ++i) {
            const unsigned short* ap = &sA[buf][(wm * 32 + i * 16 + laneR) * LDA];
#pragma unroll
            for (int e = 0; e < 8; ++e) {
                int kof = (e < 4) ? (laneH * 8 + 2 * e) : (16 + laneH * 8 + 2 * (e - 4));
                af[i].u[e] = *(const unsigned int*)&ap[kof];
            }
        }
#pragma unroll
        for (int j = 0; j < 2; ++j) {
            const unsigned short* bp = &sB[buf][(wn * 32 + j * 16 + laneR) * LDB];
#pragma unroll
            for (int e = 0; e < 8; ++e) {
                int kof = (e < 4) ? (laneH * 8 + 2 * e) : (16 + laneH * 8 + 2 * (e - 4));
                bf2[j].u[e] = *(const unsigned int*)&bp[kof];
            }
        }
#pragma unroll
        for (int i = 0; i < 2; ++i)
#pragma unroll
            for (int j = 0; j < 2; ++j)
                c[i][j] = __builtin_amdgcn_wmma_f32_16x16x32_bf16(
                    false, af[i].v, false, bf2[j].v,
                    (short)0, c[i][j], false, false);
        __syncthreads();
    }

    // ---- epilogue (C/D layout: row = r + 8*laneH, col = laneR)
#pragma unroll
    for (int i = 0; i < 2; ++i) {
#pragma unroll
        for (int j = 0; j < 2; ++j) {
            int colg = bn0 + wn * 32 + j * 16 + laneR;
            float bv = (MODE >= 1) ? bias[colg] : 0.0f;
#pragma unroll
            for (int r = 0; r < 8; ++r) {
                int rowg = bm0 + wm * 32 + i * 16 + r + 8 * laneH;
                float val = c[i][j][r] + bv;
                if (MODE == 1) {
                    val = val > 0.0f ? val : 0.0f;
                    ((unsigned short*)Cout)[(size_t)rowg * N + colg] = f32_to_bf16_rne(val);
                } else {
                    ((float*)Cout)[(size_t)rowg * N + colg] = val;
                }
            }
        }
    }
}

// -------------------------------------------------------------------- phi ---
// out[b,h,s,m] = exp(proj - 0.5*||x||^2 - max_m) / sqrt(M) + 1e-6
__global__ __launch_bounds__(128)
void phi_kernel(const float* __restrict__ X,      // [B,S,D] head-sliced q or k
                const float* __restrict__ omega,  // [M,dh] fp32
                float* __restrict__ out,          // [B*H, S, M]
                float scale)
{
    __shared__ float xs[DHc];
    __shared__ float red[Mc];
    const int s  = blockIdx.x;
    const int bh = blockIdx.y;
    const int b  = bh / Hc, h = bh % Hc;
    const int t  = threadIdx.x;

    const float* xrow = X + ((size_t)(b * Sc + s) * Dc + h * DHc);
    if (t < DHc) xs[t] = xrow[t] * scale;
    __syncthreads();

    red[t] = (t < DHc) ? xs[t] * xs[t] : 0.0f;
    __syncthreads();
    for (int off = 64; off > 0; off >>= 1) {
        if (t < off) red[t] += red[t + off];
        __syncthreads();
    }
    float xx = red[0];
    __syncthreads();

    const float* om = omega + (size_t)t * DHc;
    float proj = 0.0f;
#pragma unroll 8
    for (int d = 0; d < DHc; ++d) proj += xs[d] * om[d];
    float z = proj - 0.5f * xx;

    red[t] = z;
    __syncthreads();
    for (int off = 64; off > 0; off >>= 1) {
        if (t < off) red[t] = fmaxf(red[t], red[t + off]);
        __syncthreads();
    }
    float zmax = red[0];

    out[((size_t)bh * Sc + s) * Mc + t] =
        expf(z - zmax) * 0.08838834764831845f /* 1/sqrt(128) */ + 1e-6f;
}

// ------------------------------------------------------------------- scan ---
// Causal linear attention: one block per (b,h); S_kv[64][128] lives in LDS.
__global__ __launch_bounds__(256)
void scan_kernel(const float* __restrict__ qp,   // [B*H, S, M]
                 const float* __restrict__ kp,   // [B*H, S, M]
                 const float* __restrict__ V,    // [B,S,D]
                 float* __restrict__ ctx)        // [B,S,D]
{
    constexpr int LDSK = Mc + 1;                 // 129: odd stride, no bank conflict
    __shared__ float Skv[DHc * LDSK];
    __shared__ float sk[Mc];
    __shared__ float qs[Mc], ksm[Mc], vs[DHc];
    __shared__ float numpart[256];

    const int t  = threadIdx.x;
    const int bh = blockIdx.x;
    const int b  = bh / Hc, h = bh % Hc;
    const int d  = t & 63, qm = t >> 6;          // t = qm*64 + d

    for (int i = t; i < DHc * LDSK; i += 256) Skv[i] = 0.0f;
    if (t < Mc) sk[t] = 0.0f;
    __syncthreads();

    const float* qpb = qp + (size_t)bh * Sc * Mc;
    const float* kpb = kp + (size_t)bh * Sc * Mc;

    for (int s = 0; s < Sc; ++s) {
        if (t < Mc) { qs[t] = qpb[(size_t)s * Mc + t]; ksm[t] = kpb[(size_t)s * Mc + t]; }
        if (t >= 128 && t < 192) vs[t - 128] = V[((size_t)(b * Sc + s)) * Dc + h * DHc + (t - 128)];
        __syncthreads();

        const float vd = vs[d];
        float acc = 0.0f;
        const int m0 = qm * 32;
#pragma unroll 8
        for (int mm = 0; mm < 32; ++mm) {
            int m = m0 + mm;
            float x = Skv[d * LDSK + m] + ksm[m] * vd;
            Skv[d * LDSK + m] = x;
            acc += qs[m] * x;
        }
        numpart[t] = acc;
        if (t < Mc) sk[t] += ksm[t];
        __syncthreads();

        if (t < DHc) {
            float num = numpart[t] + numpart[t + 64] + numpart[t + 128] + numpart[t + 192];
            float den = 0.0f;
#pragma unroll 8
            for (int m = 0; m < Mc; ++m) den += qs[m] * sk[m];   // LDS broadcast reads
            ctx[((size_t)(b * Sc + s)) * Dc + h * DHc + t] = num / (den + 1e-6f);
        }
        __syncthreads();
    }
}

// -------------------------------------------------------------- layernorm ---
__global__ __launch_bounds__(256)
void ln_kernel(const float* __restrict__ x0, const float* __restrict__ attn,
               const float* __restrict__ gamma, const float* __restrict__ beta,
               unsigned short* __restrict__ hout)
{
    __shared__ float r1[256], r2[256];
    const int row = blockIdx.x;
    const int t   = threadIdx.x;
    const float* a  = x0   + (size_t)row * Dc;
    const float* at = attn + (size_t)row * Dc;

    float s1 = 0.0f, s2 = 0.0f;
    for (int i = t; i < Dc; i += 256) {
        float v = a[i] + at[i];
        s1 += v; s2 += v * v;
    }
    r1[t] = s1; r2[t] = s2;
    __syncthreads();
    for (int off = 128; off > 0; off >>= 1) {
        if (t < off) { r1[t] += r1[t + off]; r2[t] += r2[t + off]; }
        __syncthreads();
    }
    float mu  = r1[0] * (1.0f / Dc);
    float var = r2[0] * (1.0f / Dc) - mu * mu;
    float inv = rsqrtf(var + 1e6f);   // faithful to source epsilon
    for (int i = t; i < Dc; i += 256) {
        float v = a[i] + at[i];
        hout[(size_t)row * Dc + i] = f32_to_bf16_rne(gamma[i] * (v - mu) * inv + beta[i]);
    }
}

// ------------------------------------------------------------------ launch ---
extern "C" void kernel_launch(void* const* d_in, const int* in_sizes, int n_in,
                              void* d_out, int out_size, void* d_ws, size_t ws_size,
                              hipStream_t stream)
{
    const float* batch = (const float*)d_in[0];
    const float* Wq    = (const float*)d_in[1];
    const float* Wk    = (const float*)d_in[2];
    const float* Wv    = (const float*)d_in[3];
    const float* Wo    = (const float*)d_in[4];
    const float* omega = (const float*)d_in[5];
    const float* W1    = (const float*)d_in[6];
    const float* b1    = (const float*)d_in[7];
    const float* W2    = (const float*)d_in[8];
    const float* b2    = (const float*)d_in[9];
    const float* gamma = (const float*)d_in[10];
    const float* beta  = (const float*)d_in[11];

    const size_t nBSD = (size_t)Bc * Sc * Dc;          // 4,194,304
    const size_t nDD  = (size_t)Dc * Dc;               // 1,048,576
    const size_t nDF  = (size_t)Dc * DFFc;             // 4,194,304
    const size_t nBHS_M = (size_t)Bc * Hc * Sc * Mc;   // 8,388,608

    char* p = (char*)d_ws;
    auto take = [&](size_t bytes) { char* r = p; p += (bytes + 255) & ~(size_t)255; return r; };
    unsigned short* Xb  = (unsigned short*)take(nBSD * 2);  // also ctx_bf16 later
    unsigned short* Wqt = (unsigned short*)take(nDD * 2);   // transposed bf16 weights
    unsigned short* Wkt = (unsigned short*)take(nDD * 2);
    unsigned short* Wvt = (unsigned short*)take(nDD * 2);
    unsigned short* Wot = (unsigned short*)take(nDD * 2);
    unsigned short* W1t = (unsigned short*)take(nDF * 2);   // [DFF, D]
    unsigned short* W2t = (unsigned short*)take(nDF * 2);   // [D, DFF]
    float* q   = (float*)take(nBSD * 4);                    // reused as ctx
    float* k   = (float*)take(nBSD * 4);                    // reused as attn
    float* v   = (float*)take(nBSD * 4);
    float* qp  = (float*)take(nBHS_M * 4);                  // reused as ffn1 bf16
    float* kp  = (float*)take(nBHS_M * 4);
    unsigned short* hb = (unsigned short*)take(nBSD * 2);

    float* ctx  = q;                             // alias: q dead after phi
    unsigned short* ctxb = Xb;                   // alias: Xb dead after QKV GEMMs
    float* attn = k;                             // alias: k dead after phi
    unsigned short* f1b  = (unsigned short*)qp;  // alias: qp dead after scan

    const float scale = 0.3535533905932738f;     // 64^-0.25

    // 1. bf16 conversions (weights transposed: Bt[N][K])
    cvt_f32_bf16<<<(nBSD + 255) / 256, 256, 0, stream>>>(batch, Xb, (int)nBSD);
    dim3 gTdd(Dc / 32, Dc / 32);
    cvt_transpose_bf16<<<gTdd, 256, 0, stream>>>(Wq, Wqt, Dc, Dc);
    cvt_transpose_bf16<<<gTdd, 256, 0, stream>>>(Wk, Wkt, Dc, Dc);
    cvt_transpose_bf16<<<gTdd, 256, 0, stream>>>(Wv, Wvt, Dc, Dc);
    cvt_transpose_bf16<<<gTdd, 256, 0, stream>>>(Wo, Wot, Dc, Dc);
    cvt_transpose_bf16<<<dim3(DFFc / 32, Dc / 32), 256, 0, stream>>>(W1, W1t, Dc, DFFc);
    cvt_transpose_bf16<<<dim3(Dc / 32, DFFc / 32), 256, 0, stream>>>(W2, W2t, DFFc, Dc);

    // 2. QKV projections (M=4096, N=1024, K=1024)
    dim3 gP(Dc / 64, (Bc * Sc) / 128);
    gemm_wmma_bf16<0><<<gP, 256, 0, stream>>>(Xb, Wqt, nullptr, q, Bc * Sc, Dc, Dc);
    gemm_wmma_bf16<0><<<gP, 256, 0, stream>>>(Xb, Wkt, nullptr, k, Bc * Sc, Dc, Dc);
    gemm_wmma_bf16<0><<<gP, 256, 0, stream>>>(Xb, Wvt, nullptr, v, Bc * Sc, Dc, Dc);

    // 3. FAVOR+ features
    dim3 gPhi(Sc, Bc * Hc);
    phi_kernel<<<gPhi, 128, 0, stream>>>(q, omega, qp, scale);
    phi_kernel<<<gPhi, 128, 0, stream>>>(k, omega, kp, scale);

    // 4. causal linear-attention scan (one block per (b,h))
    scan_kernel<<<Bc * Hc, 256, 0, stream>>>(qp, kp, v, ctx);

    // 5. output projection
    cvt_f32_bf16<<<(nBSD + 255) / 256, 256, 0, stream>>>(ctx, ctxb, (int)nBSD);
    gemm_wmma_bf16<0><<<gP, 256, 0, stream>>>(ctxb, Wot, nullptr, attn, Bc * Sc, Dc, Dc);

    // 6. residual + layernorm -> bf16
    ln_kernel<<<Bc * Sc, 256, 0, stream>>>(batch, attn, gamma, beta, hb);

    // 7. FFN1: relu(h@W1 + b1) -> bf16   (M=4096, N=4096, K=1024)
    dim3 gF1(DFFc / 64, (Bc * Sc) / 128);
    gemm_wmma_bf16<1><<<gF1, 256, 0, stream>>>(hb, W1t, b1, f1b, Bc * Sc, DFFc, Dc);

    // 8. FFN2: f1@W2 + b2 -> fp32 out   (M=4096, N=1024, K=4096)
    gemm_wmma_bf16<2><<<gP, 256, 0, stream>>>(f1b, W2t, b2, d_out, Bc * Sc, Dc, DFFc);
}
